// PhyloLM_55164559950475
// MI455X (gfx1250) — matrix operand order repaired and checked by reading, not compiled, not run on hardware
//
#include <hip/hip_runtime.h>
#include <hip/hip_bf16.h>
#include <math.h>

typedef __bf16 bf16;
typedef __attribute__((ext_vector_type(16))) __bf16 v16bf;
typedef __attribute__((ext_vector_type(8)))  float  v8f;

#define NSEQ  24
#define NPAIR 276
#define CCOL  128
#define HH    256
#define NBLK  2
#define NHEAD 8
#define HD    32
#define DFF   1024
#define NROWS (NPAIR * CCOL)   // 35328

// low 32 bits of a flat pointer to __shared__ == LDS byte offset
__device__ __forceinline__ unsigned lds_off(const void* p) {
    return (unsigned)(unsigned long long)p;
}

// ---------------------------------------------------------------------------
// h[p,c,:] = sum_r pm[p,r] * emb[x[0,r,c], :]
// ---------------------------------------------------------------------------
__global__ void embed_pairs(const int* __restrict__ x, const float* __restrict__ pm,
                            const float* __restrict__ emb, float* __restrict__ h) {
    int p = blockIdx.x, c = blockIdx.y, t = threadIdx.x;   // blockDim.x == HH
    float acc = 0.f;
    for (int r = 0; r < NSEQ; ++r) {
        float w = pm[p * NSEQ + r];
        if (w != 0.f) {
            int tok = x[r * CCOL + c];
            acc += w * emb[(size_t)tok * HH + t];
        }
    }
    h[((size_t)p * CCOL + c) * HH + t] = acc;
}

// ---------------------------------------------------------------------------
// LayerNorm (eps 1e-5, biased variance) -> bf16, one wave per row of H=256
// ---------------------------------------------------------------------------
__global__ void layernorm_bf16(const float* __restrict__ h, const float* __restrict__ g,
                               const float* __restrict__ b, bf16* __restrict__ z) {
    int row = blockIdx.x, l = threadIdx.x;   // blockDim.x == 32
    const float* xr = h + (size_t)row * HH;
    __shared__ float red[32];
    __shared__ float stat[2];
    float vals[8];
    float s = 0.f;
    for (int i = 0; i < 8; ++i) { vals[i] = xr[l * 8 + i]; s += vals[i]; }
    red[l] = s; __syncthreads();
    if (l == 0) { float t = 0.f; for (int i = 0; i < 32; ++i) t += red[i]; stat[0] = t / (float)HH; }
    __syncthreads();
    float mean = stat[0];
    float vs = 0.f;
    for (int i = 0; i < 8; ++i) { float d = vals[i] - mean; vs += d * d; }
    red[l] = vs; __syncthreads();
    if (l == 0) { float t = 0.f; for (int i = 0; i < 32; ++i) t += red[i]; stat[1] = t / (float)HH; }
    __syncthreads();
    float rstd = rsqrtf(stat[1] + 1e-5f);
    bf16* zr = z + (size_t)row * HH;
    for (int i = 0; i < 8; ++i) {
        int c = l * 8 + i;
        zr[c] = (bf16)((vals[i] - mean) * rstd * g[c] + b[c]);
    }
}

// ---------------------------------------------------------------------------
// WMMA GEMM: out[M,N] = epi(A[M,K](bf16,row-major) @ Wt[N,K](bf16,TRANSPOSED))
// Block: 128 threads / 4 waves; tile 128x64; wave = 32 rows x 64 cols.
// A tiles DMA'd to LDS with double buffering: tile k+1 streams in via
// global_load_async_to_lds_b128 while the 8 WMMAs of tile k execute;
// s_wait_asynccnt 4 (in-order completion) releases tile k only.
// A tiles are stored in natural k order (k = 16h + s, same as B)
// in LDS, so an A fragment is one aligned 32B LDS read.
// B fragments are contiguous 32B global reads from transposed weights
// (<5MB total, permanently L2-resident).
// ---------------------------------------------------------------------------
template <bool BIAS, bool GELU, bool RESID, bool OUTBF16>
__global__ __launch_bounds__(128)
void gemm_wmma(const bf16* __restrict__ A, const bf16* __restrict__ Wt,
               const float* __restrict__ bias, const float* __restrict__ resid,
               void* __restrict__ out, int M, int N, int K) {
    __shared__ alignas(64) bf16 As[2][128][32];
    int tid = threadIdx.x;
    int wave = tid >> 5, lane = tid & 31;
    int m0 = blockIdx.x * 128, n0 = blockIdx.y * 64;

    v8f zero = {};
    v8f acc[2][4];
    for (int mi = 0; mi < 2; ++mi)
        for (int nt = 0; nt < 4; ++nt) acc[mi][nt] = zero;

    unsigned as_base = lds_off(&As[0][0][0]);
    int half16 = (lane >= 16) ? 16 : 0;
    int row_a  = wave * 32 + (lane & 15);

    // issue async DMA of one 128x32 A tile into LDS buffer `buf` (natural k order)
    auto issue_tile = [&](int k0, int buf) {
        for (int i = 0; i < 4; ++i) {
            int bid = tid + 128 * i;           // 512 16B blocks, 4 per thread
            int m   = bid >> 2;
            int kb  = bid & 3;
            int kbp = kb;
            unsigned long long src =
                (unsigned long long)(const void*)(A + (size_t)(m0 + m) * K + k0 + kb * 8);
            unsigned dst = as_base + (unsigned)(buf * 8192 + m * 64 + kbp * 16);
            asm volatile("global_load_async_to_lds_b128 %0, %1, off"
                         :: "v"(dst), "v"(src) : "memory");
        }
    };

    issue_tile(0, 0);
    for (int k0 = 0; k0 < K; k0 += 32) {
        int cur = (k0 >> 5) & 1;
        if (k0 + 32 < K) {
            issue_tile(k0 + 32, cur ^ 1);
            // async completes in order: <=4 outstanding => tile `cur` landed
            asm volatile("s_wait_asynccnt 0x4" ::: "memory");
        } else {
            asm volatile("s_wait_asynccnt 0x0" ::: "memory");
        }
        if (k0 + 64 < K)
            __builtin_prefetch((const void*)(A + (size_t)(m0 + tid) * K + k0 + 64), 0, 1);
        __syncthreads();

        v16bf a0 = *(const v16bf*)&As[cur][row_a][half16];
        v16bf a1 = *(const v16bf*)&As[cur][row_a + 16][half16];
        for (int nt = 0; nt < 4; ++nt) {
            int col = n0 + nt * 16 + (lane & 15);
            v16bf bm = *(const v16bf*)(Wt + (size_t)col * K + k0 + half16);
            acc[0][nt] = __builtin_amdgcn_wmma_f32_16x16x32_bf16(
                false, a0, false, bm, (short)0, acc[0][nt], false, false);
            acc[1][nt] = __builtin_amdgcn_wmma_f32_16x16x32_bf16(
                false, a1, false, bm, (short)0, acc[1][nt], false, false);
        }
        __syncthreads();   // all waves done with buf `cur` before it is re-DMA'd
    }

    int col_l = lane & 15;
    for (int mi = 0; mi < 2; ++mi) {
        int rbase = m0 + wave * 32 + mi * 16 + ((lane >= 16) ? 8 : 0);
        for (int nt = 0; nt < 4; ++nt) {
            int col = n0 + nt * 16 + col_l;
            float bv = BIAS ? bias[col] : 0.f;
            for (int g = 0; g < 8; ++g) {
                int row = rbase + g;
                float v = acc[mi][nt][g] + bv;
                if (GELU) v = 0.5f * v * (1.f + erff(v * 0.70710678118654752f));
                if (RESID) v += resid[(size_t)row * N + col];
                if (OUTBF16) ((bf16*)out)[(size_t)row * N + col] = (bf16)v;
                else         ((float*)out)[(size_t)row * N + col] = v;
            }
        }
    }
}

// ---------------------------------------------------------------------------
// Fused attention: one wave per (batch, head, 16-row q tile).
// Row offset (elements) of position s in batch b: b*bStride + s*sStride.
// SPAD = sequence length padded to a multiple of 32 (compile-time: LDS sizing).
// ---------------------------------------------------------------------------
template <int SPAD>
__global__ __launch_bounds__(32)
void attn_wmma(const bf16* __restrict__ qb, const bf16* __restrict__ kb,
               const bf16* __restrict__ vb, bf16* __restrict__ ob,
               int S, long bStride, long sStride) {
    int batch = blockIdx.x, head = blockIdx.y, qt = blockIdx.z;
    int lane = threadIdx.x;

    __shared__ alignas(64) float sc[16][SPAD];
    __shared__ alignas(64) bf16  pbuf[16][SPAD];   // natural k order within each 32-chunk
    __shared__ alignas(64) bf16  vs_t[32][32];     // V chunk, transposed [d][k]

    long hoff = (long)head * HD;
    int kbA    = (lane >= 16) ? 16 : 0;
    int half16 = (lane >= 16) ? 16 : 0;

    // Q tile in A-layout (two 16B groups; compiler vectorizes)
    int qrow_l = qt * 16 + (lane & 15);
    int qrow_c = (qrow_l < S) ? qrow_l : (S - 1);
    long qoff = (long)batch * bStride + (long)qrow_c * sStride + hoff;
    v16bf qa;
    for (int e = 0; e < 8; ++e)  qa[e] = qb[qoff + kbA + e];
    for (int e = 8; e < 16; ++e) qa[e] = qb[qoff + kbA + e];

    const float scale = 0.17677669529663687f;   // 1/sqrt(32)

    // scores = Q @ K^T : B operand column n is K row n -> contiguous 32B load
    for (int kt = 0; kt < SPAD / 16; ++kt) {
        int krow = kt * 16 + (lane & 15);
        v16bf km;
        if (krow < S) {
            long koff = (long)batch * bStride + (long)krow * sStride + hoff;
            km = *(const v16bf*)(kb + koff + half16);
        } else {
            for (int e = 0; e < 16; ++e) km[e] = (bf16)0.f;
        }
        v8f c = {};
        c = __builtin_amdgcn_wmma_f32_16x16x32_bf16(false, qa, false, km, (short)0, c, false, false);
        int rr = (lane >= 16) ? 8 : 0;
        for (int g = 0; g < 8; ++g)
            sc[rr + g][kt * 16 + (lane & 15)] = c[g] * scale;
    }
    __syncthreads();

    // softmax per q row; write P in natural k order within each 32-chunk,
    // so A fragments are contiguous (k = 16h + s).
    if (lane < 16) {
        float m = -1e30f;
        for (int c = 0; c < S; ++c) m = fmaxf(m, sc[lane][c]);
        float s = 0.f;
        for (int c = 0; c < S; ++c) { float e = __expf(sc[lane][c] - m); sc[lane][c] = e; s += e; }
        float inv = 1.f / s;
        for (int c = 0; c < SPAD; ++c) {
            int j = c & 31;
            int jp = j;
            pbuf[lane][(c & ~31) + jp] = (c < S) ? (bf16)(sc[lane][c] * inv) : (bf16)0.f;
        }
    }
    __syncthreads();

    // O = P @ V  (d = 32 -> two N tiles per 32-key chunk)
    v8f o0 = {}, o1 = {};
    for (int ch = 0; ch < SPAD / 32; ++ch) {
        // stage V chunk transposed: lane owns key row s = ch*32+lane
        {
            int s = ch * 32 + lane;
            if (s < S) {
                long voff = (long)batch * bStride + (long)s * sStride + hoff;
                for (int d = 0; d < 32; ++d) vs_t[d][lane] = vb[voff + d];
            } else {
                for (int d = 0; d < 32; ++d) vs_t[d][lane] = (bf16)0.f;
            }
        }
        __syncthreads();   // single-wave workgroup: lowers to S_NOP
        v16bf pa = *(const v16bf*)&pbuf[lane & 15][ch * 32 + half16];
        for (int dt = 0; dt < 2; ++dt) {
            int dcol = dt * 16 + (lane & 15);
            v16bf vm = *(const v16bf*)&vs_t[dcol][half16];
            if (dt == 0) o0 = __builtin_amdgcn_wmma_f32_16x16x32_bf16(false, pa, false, vm, (short)0, o0, false, false);
            else         o1 = __builtin_amdgcn_wmma_f32_16x16x32_bf16(false, pa, false, vm, (short)0, o1, false, false);
        }
        __syncthreads();
    }

    int rr = (lane >= 16) ? 8 : 0;
    for (int g = 0; g < 8; ++g) {
        int qrow = qt * 16 + rr + g;
        if (qrow < S) {
            long ooff = (long)batch * bStride + (long)qrow * sStride + hoff;
            ob[ooff + (lane & 15)]      = (bf16)o0[g];
            ob[ooff + 16 + (lane & 15)] = (bf16)o1[g];
        }
    }
}

// ---------------------------------------------------------------------------
__global__ void cast_f32_bf16(const float* __restrict__ src, bf16* __restrict__ dst, long n) {
    long i = (long)blockIdx.x * blockDim.x + threadIdx.x;
    if (i < n) dst[i] = (bf16)src[i];
}

// transpose-convert weights: src f32 [K,N] -> dst bf16 [N,K]
__global__ void convt_f32_bf16(const float* __restrict__ src, bf16* __restrict__ dst,
                               int K, int N) {
    long i = (long)blockIdx.x * blockDim.x + threadIdx.x;
    if (i < (long)K * N) {
        int k = (int)(i / N), n = (int)(i % N);
        dst[(size_t)n * K + k] = (bf16)src[i];
    }
}

// y[row] = sum_f g[row,f]*w[f] + b
__global__ void rowdot(const bf16* __restrict__ g, const float* __restrict__ w,
                       const float* __restrict__ b, float* __restrict__ y) {
    int row = blockIdx.x, l = threadIdx.x;   // blockDim.x == 32
    __shared__ float red[32];
    const bf16* gr = g + (size_t)row * DFF;
    float s = 0.f;
    for (int i = l; i < DFF; i += 32) s += (float)gr[i] * w[i];
    red[l] = s; __syncthreads();
    if (l == 0) { float t = 0.f; for (int i = 0; i < 32; ++i) t += red[i]; y[row] = t + b[0]; }
}

// out[p] = sum_c y[p,c]*uw[c] + ub
__global__ void pair_reduce(const float* __restrict__ y, const float* __restrict__ uw,
                            const float* __restrict__ ub, float* __restrict__ out) {
    int p = blockIdx.x, t = threadIdx.x;   // blockDim.x == CCOL
    __shared__ float red[CCOL];
    red[t] = y[(size_t)p * CCOL + t] * uw[t];
    __syncthreads();
    for (int s = CCOL / 2; s > 0; s >>= 1) {
        if (t < s) red[t] += red[t + s];
        __syncthreads();
    }
    if (t == 0) out[p] = red[0] + ub[0];
}

// ---------------------------------------------------------------------------
extern "C" void kernel_launch(void* const* d_in, const int* in_sizes, int n_in,
                              void* d_out, int out_size, void* d_ws, size_t ws_size,
                              hipStream_t stream) {
    (void)in_sizes; (void)n_in; (void)out_size; (void)ws_size;
    const int*   x    = (const int*)  d_in[0];
    const float* pm   = (const float*)d_in[1];
    const float* emb  = (const float*)d_in[2];
    const float* ln1g = (const float*)d_in[3];
    const float* ln1b = (const float*)d_in[4];
    const float* wqr  = (const float*)d_in[5];
    const float* wkr  = (const float*)d_in[6];
    const float* wvr  = (const float*)d_in[7];
    const float* wor  = (const float*)d_in[8];
    const float* ln2g = (const float*)d_in[9];
    const float* ln2b = (const float*)d_in[10];
    const float* wqc  = (const float*)d_in[11];
    const float* wkc  = (const float*)d_in[12];
    const float* wvc  = (const float*)d_in[13];
    const float* woc  = (const float*)d_in[14];
    const float* ln3g = (const float*)d_in[15];
    const float* ln3b = (const float*)d_in[16];
    const float* fw1  = (const float*)d_in[17];
    const float* fb1  = (const float*)d_in[18];
    const float* fw2  = (const float*)d_in[19];
    const float* fb2  = (const float*)d_in[20];
    const float* pw1  = (const float*)d_in[21];
    const float* pb1  = (const float*)d_in[22];
    const float* pw2  = (const float*)d_in[23];
    const float* pb2  = (const float*)d_in[24];
    const float* uw   = (const float*)d_in[25];
    const float* ub   = (const float*)d_in[26];

    // ---- workspace layout ----
    char* ws = (char*)d_ws;
    float* h = (float*)ws;                                        // N*H f32
    size_t off = (size_t)NROWS * HH * 4;
    bf16* z = (bf16*)(ws + off);                                  // N*H bf16
    off += (size_t)NROWS * HH * 2;
    bf16* q = (bf16*)(ws + off);                                  // q/k/v/o, aliased by mid
    bf16* k = q + (size_t)NROWS * HH;
    bf16* v = k + (size_t)NROWS * HH;
    bf16* o = v + (size_t)NROWS * HH;
    bf16* mid = q;                                                // N*DFF bf16 (overlaps q..o)
    off += (size_t)NROWS * DFF * 2;
    float* y = (float*)(ws + off);                                // N f32
    off += (size_t)NROWS * 4;
    bf16* wb = (bf16*)(ws + off);                                 // transposed bf16 weights

    bf16* wqr_b = wb;
    bf16* wkr_b = wqr_b + (size_t)NBLK * HH * HH;
    bf16* wvr_b = wkr_b + (size_t)NBLK * HH * HH;
    bf16* wor_b = wvr_b + (size_t)NBLK * HH * HH;
    bf16* wqc_b = wor_b + (size_t)NBLK * HH * HH;
    bf16* wkc_b = wqc_b + (size_t)NBLK * HH * HH;
    bf16* wvc_b = wkc_b + (size_t)NBLK * HH * HH;
    bf16* woc_b = wvc_b + (size_t)NBLK * HH * HH;
    bf16* fw1_b = woc_b + (size_t)NBLK * HH * HH;
    bf16* fw2_b = fw1_b + (size_t)NBLK * HH * DFF;
    bf16* pw1_b = fw2_b + (size_t)NBLK * DFF * HH;

    auto convt = [&](const float* src, bf16* dst, int K, int N) {
        long n = (long)K * N;
        convt_f32_bf16<<<dim3((unsigned)((n + 255) / 256)), 256, 0, stream>>>(src, dst, K, N);
    };
    for (int i = 0; i < NBLK; ++i) {
        convt(wqr + (size_t)i * HH * HH, wqr_b + (size_t)i * HH * HH, HH, HH);
        convt(wkr + (size_t)i * HH * HH, wkr_b + (size_t)i * HH * HH, HH, HH);
        convt(wvr + (size_t)i * HH * HH, wvr_b + (size_t)i * HH * HH, HH, HH);
        convt(wor + (size_t)i * HH * HH, wor_b + (size_t)i * HH * HH, HH, HH);
        convt(wqc + (size_t)i * HH * HH, wqc_b + (size_t)i * HH * HH, HH, HH);
        convt(wkc + (size_t)i * HH * HH, wkc_b + (size_t)i * HH * HH, HH, HH);
        convt(wvc + (size_t)i * HH * HH, wvc_b + (size_t)i * HH * HH, HH, HH);
        convt(woc + (size_t)i * HH * HH, woc_b + (size_t)i * HH * HH, HH, HH);
        convt(fw1 + (size_t)i * HH * DFF, fw1_b + (size_t)i * HH * DFF, HH, DFF);
        convt(fw2 + (size_t)i * DFF * HH, fw2_b + (size_t)i * DFF * HH, DFF, HH);
    }
    convt(pw1, pw1_b, HH, DFF);

    // ---- embedding + pair gather ----
    embed_pairs<<<dim3(NPAIR, CCOL), HH, 0, stream>>>(x, pm, emb, h);

    const dim3 gH(NROWS / 128, HH / 64);    // 276 x 4
    const dim3 gF(NROWS / 128, DFF / 64);   // 276 x 16

    for (int i = 0; i < NBLK; ++i) {
        // ---- row attention (along C=128, per pair) ----
        layernorm_bf16<<<NROWS, 32, 0, stream>>>(h, ln1g + i * HH, ln1b + i * HH, z);
        gemm_wmma<false, false, false, true><<<gH, 128, 0, stream>>>(z, wqr_b + (size_t)i * HH * HH, nullptr, nullptr, q, NROWS, HH, HH);
        gemm_wmma<false, false, false, true><<<gH, 128, 0, stream>>>(z, wkr_b + (size_t)i * HH * HH, nullptr, nullptr, k, NROWS, HH, HH);
        gemm_wmma<false, false, false, true><<<gH, 128, 0, stream>>>(z, wvr_b + (size_t)i * HH * HH, nullptr, nullptr, v, NROWS, HH, HH);
        attn_wmma<128><<<dim3(NPAIR, NHEAD, CCOL / 16), 32, 0, stream>>>(q, k, v, o, CCOL, (long)CCOL * HH, (long)HH);
        gemm_wmma<false, false, true, false><<<gH, 128, 0, stream>>>(o, wor_b + (size_t)i * HH * HH, nullptr, h, h, NROWS, HH, HH);

        // ---- column attention (along P=276, per column) ----
        layernorm_bf16<<<NROWS, 32, 0, stream>>>(h, ln2g + i * HH, ln2b + i * HH, z);
        gemm_wmma<false, false, false, true><<<gH, 128, 0, stream>>>(z, wqc_b + (size_t)i * HH * HH, nullptr, nullptr, q, NROWS, HH, HH);
        gemm_wmma<false, false, false, true><<<gH, 128, 0, stream>>>(z, wkc_b + (size_t)i * HH * HH, nullptr, nullptr, k, NROWS, HH, HH);
        gemm_wmma<false, false, false, true><<<gH, 128, 0, stream>>>(z, wvc_b + (size_t)i * HH * HH, nullptr, nullptr, v, NROWS, HH, HH);
        attn_wmma<288><<<dim3(CCOL, NHEAD, (NPAIR + 15) / 16), 32, 0, stream>>>(q, k, v, o, NPAIR, (long)HH, (long)CCOL * HH);
        gemm_wmma<false, false, true, false><<<gH, 128, 0, stream>>>(o, woc_b + (size_t)i * HH * HH, nullptr, h, h, NROWS, HH, HH);

        // ---- feed-forward ----
        layernorm_bf16<<<NROWS, 32, 0, stream>>>(h, ln3g + i * HH, ln3b + i * HH, z);
        gemm_wmma<true, true, false, true><<<gF, 128, 0, stream>>>(z, fw1_b + (size_t)i * HH * DFF, fb1 + i * DFF, nullptr, mid, NROWS, DFF, HH);
        gemm_wmma<true, false, true, false><<<gH, 128, 0, stream>>>(mid, fw2_b + (size_t)i * DFF * HH, fb2 + i * HH, h, h, NROWS, HH, DFF);
    }

    // ---- output head ----
    cast_f32_bf16<<<dim3((unsigned)(((long)NROWS * HH + 255) / 256)), 256, 0, stream>>>(h, z, (long)NROWS * HH);
    gemm_wmma<true, true, false, true><<<gF, 128, 0, stream>>>(z, pw1_b, pb1, nullptr, mid, NROWS, DFF, HH);
    rowdot<<<NROWS, 32, 0, stream>>>(mid, pw2, pb2, y);
    pair_reduce<<<NPAIR, CCOL, 0, stream>>>(y, uw, ub, (float*)d_out);
}
